// GeneralAttn_88184268522171
// MI455X (gfx1250) — matrix operand order repaired and checked by reading, not compiled, not run on hardware
//
#include <hip/hip_runtime.h>

// Problem constants (reference): B=4, NP=1024, E=1024, H=16, D=64
constexpr int kB  = 4;
constexpr int kNP = 1024;
constexpr int kE  = 1024;
constexpr int kH  = 16;
constexpr int kD  = 64;
constexpr int k3E = 3 * kE;
constexpr float kScale = 0.125f;  // 1/sqrt(64)

typedef __attribute__((ext_vector_type(16))) _Float16 v16h;
typedef __attribute__((ext_vector_type(8)))  _Float16 v8h;
typedef __attribute__((ext_vector_type(8)))  float    v8f;

#define WMMA_F16(a, b, c) \
  __builtin_amdgcn_wmma_f32_16x16x32_f16(false, (a), false, (b), (short)0, (c), false, false)

// ---------------- fragment loaders (ISA 7.12.2 layouts, wave32) ----------------
// A-frag (16x32 f16): lanes 0-15 -> row L, K={0..7,16..23}; lanes 16-31 -> row L-16, K={8..15,24..31}
__device__ inline v16h ldsA_frag(const _Float16* __restrict__ tile, int ld) {
  const int lane = threadIdx.x & 31;
  const int row  = lane & 15;
  const int kb   = (lane & 16) >> 1;  // 0 or 8
  const _Float16* p = tile + row * ld + kb;
  v8h lo = *(const v8h*)(p);
  v8h hi = *(const v8h*)(p + 16);
  v16h f;
#pragma unroll
  for (int i = 0; i < 8; ++i) { f[i] = lo[i]; f[i + 8] = hi[i]; }
  return f;
}

// B-frag (32x16 f16): lanes 0-15 -> col L, K={0..7,16..23}; lanes 16-31 -> col L-16, K={8..15,24..31}
__device__ inline v16h ldsB_frag(const _Float16* __restrict__ tile, int ld) {
  const int lane = threadIdx.x & 31;
  const int col  = lane & 15;
  const int kb   = (lane & 16) >> 1;  // 0 or 8
  const _Float16* p = tile + col * ld + kb;
  v8h lo = *(const v8h*)(p);
  v8h hi = *(const v8h*)(p + 16);
  v16h f;
#pragma unroll
  for (int i = 0; i < 8; ++i) { f[i] = lo[i]; f[i + 8] = hi[i]; }
  return f;
}

// A-frag straight from row-major fp32 global memory (f32 -> f16 convert)
__device__ inline v16h gmemA_frag(const float* __restrict__ tile, int ld) {
  const int lane = threadIdx.x & 31;
  const int row  = lane & 15;
  const int kb   = (lane & 16) >> 1;
  const float* p = tile + (size_t)row * ld + kb;
  v16h f;
#pragma unroll
  for (int i = 0; i < 8; ++i) { f[i] = (_Float16)p[i]; f[i + 8] = (_Float16)p[i + 16]; }
  return f;
}

// ---------------- GEMM: C[M,N] = A[M,K] @ W[N,K]^T + bias[N] ----------------
// 64x64 block tile, K-step 64: 8 WMMAs per wave between barriers.
constexpr int GL = 72;  // padded LDS stride (f16) for 64-wide K tiles: 144B rows, 16B aligned

__global__ __launch_bounds__(128) void gemm_xwT_bias_f16wmma(
    const float* __restrict__ A, const float* __restrict__ W,
    const float* __restrict__ bias, float* __restrict__ C,
    int M, int N, int K) {
  __shared__ alignas(16) _Float16 lA[64 * GL];
  __shared__ alignas(16) _Float16 lB[64 * GL];

  const int tid  = threadIdx.x;
  const int lane = tid & 31;
  const int wave = tid >> 5;
  const int wr   = (wave >> 1) * 32;   // wave's row offset inside 64x64 block
  const int wc   = (wave & 1) * 32;    // wave's col offset
  const size_t m0 = (size_t)blockIdx.y * 64;
  const size_t n0 = (size_t)blockIdx.x * 64;

  v8f acc[2][2];
#pragma unroll
  for (int i = 0; i < 2; ++i)
#pragma unroll
    for (int j = 0; j < 2; ++j)
#pragma unroll
      for (int r = 0; r < 8; ++r) acc[i][j][r] = 0.0f;

  const int sr = tid >> 1;            // staging row 0..63
  const int sc = (tid & 1) * 32;      // staging col 0 or 32
  const float* As = A + (m0 + sr) * (size_t)K + sc;
  const float* Ws = W + (n0 + sr) * (size_t)K + sc;
  _Float16* Ad = lA + sr * GL + sc;
  _Float16* Wd = lB + sr * GL + sc;

  for (int k0 = 0; k0 < K; k0 += 64) {
#pragma unroll
    for (int i = 0; i < 32; ++i) Ad[i] = (_Float16)As[k0 + i];
#pragma unroll
    for (int i = 0; i < 32; ++i) Wd[i] = (_Float16)Ws[k0 + i];
    __syncthreads();
    v16h a00 = ldsA_frag(lA + (wr +  0) * GL,      GL);  // rows wr..wr+15,  k 0..31
    v16h a01 = ldsA_frag(lA + (wr +  0) * GL + 32, GL);  //                  k 32..63
    v16h a10 = ldsA_frag(lA + (wr + 16) * GL,      GL);
    v16h a11 = ldsA_frag(lA + (wr + 16) * GL + 32, GL);
    v16h b00 = ldsB_frag(lB + (wc +  0) * GL,      GL);
    v16h b01 = ldsB_frag(lB + (wc +  0) * GL + 32, GL);
    v16h b10 = ldsB_frag(lB + (wc + 16) * GL,      GL);
    v16h b11 = ldsB_frag(lB + (wc + 16) * GL + 32, GL);
    acc[0][0] = WMMA_F16(a00, b00, acc[0][0]);
    acc[0][1] = WMMA_F16(a00, b10, acc[0][1]);
    acc[1][0] = WMMA_F16(a10, b00, acc[1][0]);
    acc[1][1] = WMMA_F16(a10, b10, acc[1][1]);
    acc[0][0] = WMMA_F16(a01, b01, acc[0][0]);
    acc[0][1] = WMMA_F16(a01, b11, acc[0][1]);
    acc[1][0] = WMMA_F16(a11, b01, acc[1][0]);
    acc[1][1] = WMMA_F16(a11, b11, acc[1][1]);
    __syncthreads();
  }

  // C-frag layout: lane = col (0..15 / 16..31 shifted), VGPR r -> row r (+8 for upper lanes)
  const int col = lane & 15;
  const int rb  = (lane & 16) >> 1;
#pragma unroll
  for (int j = 0; j < 2; ++j) {
    const size_t gc = n0 + wc + j * 16 + col;
    const float bv = bias[gc];
#pragma unroll
    for (int i = 0; i < 2; ++i)
#pragma unroll
      for (int r = 0; r < 8; ++r)
        C[(m0 + wr + i * 16 + rb + r) * (size_t)N + gc] = acc[i][j][r] + bv;
  }
}

// ---------------- Flash attention with additive bias ----------------
constexpr int AL = 72;  // padded LDS stride (f16) for 64-wide tiles: 144B rows, 16B aligned

__global__ __launch_bounds__(128) void attn_f16wmma(
    const float* __restrict__ qkv,   // [B*NP, 3E]
    const float* __restrict__ bias,  // [B,H,NP,NP]
    float* __restrict__ out) {       // [B*NP, E]
  __shared__ alignas(16) _Float16 lK [64 * AL];      // keys x d, row-major
  __shared__ alignas(16) _Float16 lVt[64 * AL];      // d x keys (transposed)
  __shared__ alignas(16) _Float16 lP [4 * 16 * AL];  // per-wave P tiles

  const int tid  = threadIdx.x;
  const int lane = tid & 31;
  const int wave = tid >> 5;
  const int bh   = blockIdx.x;                 // b*16 + h
  const int b    = bh >> 4;
  const int h    = bh & 15;
  const int q0   = blockIdx.y * 64 + wave * 16;  // this wave's first query row

  const int col = lane & 15;
  const int rb  = (lane & 16) >> 1;            // 0 or 8

  // Q fragments stay resident across the whole key sweep
  const float* Qbase = qkv + ((size_t)(b * kNP + q0)) * k3E + h * kD;
  const v16h qf0 = gmemA_frag(Qbase,      k3E);  // d 0..31
  const v16h qf1 = gmemA_frag(Qbase + 32, k3E);  // d 32..63

  float mrow[8], lrow[8];
  v8f Oacc[4];
#pragma unroll
  for (int r = 0; r < 8; ++r) { mrow[r] = -3.0e38f; lrow[r] = 0.0f; }
#pragma unroll
  for (int t = 0; t < 4; ++t)
#pragma unroll
    for (int r = 0; r < 8; ++r) Oacc[t][r] = 0.0f;

  _Float16* lPw = lP + wave * 16 * AL;
  const int sr = tid >> 1;          // staging key row 0..63
  const int sc = (tid & 1) * 32;    // staging d offset 0 or 32

  for (int kb0 = 0; kb0 < kNP; kb0 += 64) {
    {  // stage K (row-major) and V (transposed) into LDS as f16
      const float* ks = qkv + ((size_t)(b * kNP + kb0 + sr)) * k3E +     kE + h * kD + sc;
      const float* vs = qkv + ((size_t)(b * kNP + kb0 + sr)) * k3E + 2 * kE + h * kD + sc;
      _Float16* kd = lK + sr * AL + sc;
#pragma unroll
      for (int i = 0; i < 32; ++i) kd[i] = (_Float16)ks[i];
#pragma unroll
      for (int i = 0; i < 32; ++i) lVt[(sc + i) * AL + sr] = (_Float16)vs[i];
    }
    __syncthreads();

    // S = Q @ K^T * scale + bias ; track running row max
    float mnew[8];
#pragma unroll
    for (int r = 0; r < 8; ++r) mnew[r] = mrow[r];
    v8f S[4];
#pragma unroll
    for (int t = 0; t < 4; ++t) {
      v16h bk0 = ldsB_frag(lK + (t * 16) * AL,      AL);
      v16h bk1 = ldsB_frag(lK + (t * 16) * AL + 32, AL);
      v8f s;
#pragma unroll
      for (int r = 0; r < 8; ++r) s[r] = 0.0f;
      s = WMMA_F16(qf0, bk0, s);
      s = WMMA_F16(qf1, bk1, s);
      const float* bp = bias + (((size_t)bh * kNP) + (q0 + rb)) * kNP + kb0 + t * 16 + col;
      __builtin_prefetch(bp + 64, 0, 0);  // next key block's bias tile
#pragma unroll
      for (int r = 0; r < 8; ++r) {
        s[r] = s[r] * kScale + bp[(size_t)r * kNP];
        mnew[r] = fmaxf(mnew[r], s[r]);
      }
      S[t] = s;
    }
    // row max across the 16 lanes holding the same rows (wave32 half-groups)
#pragma unroll
    for (int r = 0; r < 8; ++r) {
      float v = mnew[r];
#pragma unroll
      for (int off = 1; off < 16; off <<= 1) v = fmaxf(v, __shfl_xor(v, off, 32));
      mnew[r] = v;
    }
    float psum[8];
#pragma unroll
    for (int r = 0; r < 8; ++r) {
      const float f = __expf(mrow[r] - mnew[r]);
      lrow[r] *= f;
      mrow[r]  = mnew[r];
      psum[r]  = 0.0f;
#pragma unroll
      for (int t = 0; t < 4; ++t) Oacc[t][r] *= f;
    }
    // P = exp(S - m); stash in LDS to re-shape C-layout -> A-layout
#pragma unroll
    for (int t = 0; t < 4; ++t)
#pragma unroll
      for (int r = 0; r < 8; ++r) {
        const float p = __expf(S[t][r] - mnew[r]);
        psum[r] += p;
        lPw[(rb + r) * AL + t * 16 + col] = (_Float16)p;
      }
#pragma unroll
    for (int r = 0; r < 8; ++r) {
      float v = psum[r];
#pragma unroll
      for (int off = 1; off < 16; off <<= 1) v += __shfl_xor(v, off, 32);
      lrow[r] += v;
    }
    __syncthreads();

    // O += P @ V
    v16h pf0 = ldsA_frag(lPw,      AL);  // keys 0..31
    v16h pf1 = ldsA_frag(lPw + 32, AL);  // keys 32..63
#pragma unroll
    for (int td = 0; td < 4; ++td) {
      v16h bv0 = ldsB_frag(lVt + (td * 16) * AL,      AL);
      v16h bv1 = ldsB_frag(lVt + (td * 16) * AL + 32, AL);
      Oacc[td] = WMMA_F16(pf0, bv0, Oacc[td]);
      Oacc[td] = WMMA_F16(pf1, bv1, Oacc[td]);
    }
    __syncthreads();
  }

  // normalize and scatter back to [B,NP,H*D] layout
#pragma unroll
  for (int r = 0; r < 8; ++r) {
    const float inv = 1.0f / lrow[r];
    float* op = out + ((size_t)(b * kNP + q0 + rb + r)) * kE + h * kD + col;
#pragma unroll
    for (int td = 0; td < 4; ++td) op[td * 16] = Oacc[td][r] * inv;
  }
}

// ---------------- launcher ----------------
extern "C" void kernel_launch(void* const* d_in, const int* in_sizes, int n_in,
                              void* d_out, int out_size, void* d_ws, size_t ws_size,
                              hipStream_t stream) {
  (void)in_sizes; (void)n_in; (void)out_size; (void)ws_size;
  const float* x     = (const float*)d_in[0];
  const float* ab    = (const float*)d_in[1];
  const float* w_in  = (const float*)d_in[2];
  const float* b_in  = (const float*)d_in[3];
  const float* w_out = (const float*)d_in[4];
  const float* b_out = (const float*)d_in[5];
  float* out = (float*)d_out;

  // workspace: qkv [4096,3072] f32 (48MB) + attn-out [4096,1024] f32 (16MB) = 64MB
  float* qkv = (float*)d_ws;
  float* att = qkv + (size_t)(kB * kNP) * k3E;

  dim3 blk(128);
  gemm_xwT_bias_f16wmma<<<dim3(k3E / 64, (kB * kNP) / 64), blk, 0, stream>>>(
      x, w_in, b_in, qkv, kB * kNP, k3E, kE);
  attn_f16wmma<<<dim3(kB * kH, kNP / 64), blk, 0, stream>>>(qkv, ab, att);
  gemm_xwT_bias_f16wmma<<<dim3(kE / 64, (kB * kNP) / 64), blk, 0, stream>>>(
      att, w_out, b_out, out, kB * kNP, kE, kE);
}
